// LSTM_10746008174929
// MI455X (gfx1250) — compile-verified, matrix operand order checked by reading, not run on hardware
//
#include <hip/hip_runtime.h>

typedef __attribute__((ext_vector_type(16))) _Float16 v16h;
typedef __attribute__((ext_vector_type(8)))  _Float16 v8h;
typedef __attribute__((ext_vector_type(8)))  float    v8f;

#define B_TOTAL 4096
#define T_STEPS 512
#define HID 32
#define WAVES_PER_BLOCK 8
#define ROWS_PER_WAVE 16
#define TCHUNK 8

// Branch-free tanh: prefer gfx1250 V_TANH_F32, else exp2/rcp straight-line form.
__device__ __forceinline__ float fast_tanh(float x) {
#if __has_builtin(__builtin_amdgcn_tanhf)
    return __builtin_amdgcn_tanhf(x);
#else
    const float e = __builtin_amdgcn_exp2f(x * 2.8853900817779268f); // 2*log2(e)
    return 1.0f - 2.0f * __builtin_amdgcn_rcpf(e + 1.0f);
#endif
}
__device__ __forceinline__ float fast_sigmoid(float x) {
    return __builtin_fmaf(0.5f, fast_tanh(0.5f * x), 0.5f);
}

__launch_bounds__(256, 1)
__global__ void lstm_recur_kernel(const float* __restrict__ x,
                                  const float* __restrict__ W_ih,
                                  const float* __restrict__ W_hh,
                                  const float* __restrict__ b_ih,
                                  const float* __restrict__ b_hh,
                                  const float* __restrict__ W_fc,
                                  const float* __restrict__ b_fc,
                                  float* __restrict__ out) {
    // Per-wave LDS slice holding h_t as f16 in row-major [16][32].
    __shared__ __align__(64) _Float16 hl_all[WAVES_PER_BLOCK][ROWS_PER_WAVE * HID];

    const int tid    = threadIdx.x;
    const int lane   = tid & 31;
    const int wslice = tid >> 5;
    _Float16* hl = hl_all[wslice];

    const int waveGlobal = (int)(blockIdx.x * blockDim.x + tid) >> 5;
    const int rowbase    = waveGlobal * ROWS_PER_WAVE;

    const int n  = lane & 15;  // WMMA column (B/C/D) / row (A)
    const int hi = lane >> 4;  // half-wave K-split selector

    // f16 element e of a v16h maps to K(e) = e + 8*hi + (e>=8 ? 8 : 0).

    // ---- Preload B tiles: Btile[j][k][n] = W_hh[16j+n][k]  (gates = h @ W_hh^T)
    v16h btile[8];
#pragma unroll
    for (int j = 0; j < 8; ++j) {
#pragma unroll
        for (int e = 0; e < 16; ++e) {
            const int k = e + 8 * hi + ((e >= 8) ? 8 : 0);
            btile[j][e] = (_Float16)W_hh[(16 * j + n) * HID + k];
        }
    }

    // Per-lane input weight and fused bias for gate column g = 16j + n.
    float wih[8], bsum[8];
#pragma unroll
    for (int j = 0; j < 8; ++j) {
        const int g = 16 * j + n;
        wih[j]  = W_ih[g];
        bsum[j] = b_ih[g] + b_hh[g];
    }

    // h0 = 0 (wave-local slice; LDS is in-order within a wave)
    for (int i = lane; i < ROWS_PER_WAVE * HID; i += 32) hl[i] = (_Float16)0.0f;
    __builtin_amdgcn_wave_barrier();

    // Cell state c in registers, C/D layout: cst[jp*8+r] <-> (row r+8*hi, hidden 16*jp+n)
    float cst[16];
#pragma unroll
    for (int i = 0; i < 16; ++i) cst[i] = 0.0f;

    v8f acc[8];

    for (int tc = 0; tc < T_STEPS; tc += TCHUNK) {
        // ---- Bulk-load x for the next 8 serial steps: contiguous in t per row.
        float xch[8][TCHUNK];
#pragma unroll
        for (int r = 0; r < 8; ++r) {
            const int row = rowbase + r + 8 * hi;
            const float4* px = (const float4*)(x + row * T_STEPS + tc);
            const float4 x0 = px[0];
            const float4 x1 = px[1];
            xch[r][0] = x0.x; xch[r][1] = x0.y; xch[r][2] = x0.z; xch[r][3] = x0.w;
            xch[r][4] = x1.x; xch[r][5] = x1.y; xch[r][6] = x1.z; xch[r][7] = x1.w;
        }

#pragma unroll
        for (int tt = 0; tt < TCHUNK; ++tt) {
            // xg_t folded into the WMMA C operand (D layout).
#pragma unroll
            for (int j = 0; j < 8; ++j) {
#pragma unroll
                for (int r = 0; r < 8; ++r)
                    acc[j][r] = __builtin_fmaf(xch[r][tt], wih[j], bsum[j]);
            }

            // ---- A = h_{t-1}: two contiguous 16B f16 chunks per lane from LDS.
            const v8h* plo = (const v8h*)(hl + n * HID + 8 * hi);
            const v8h* phi = (const v8h*)(hl + n * HID + 16 + 8 * hi);
            const v8h alo = *plo;
            const v8h ahi = *phi;
            v16h a;
#pragma unroll
            for (int e = 0; e < 8; ++e) { a[e] = alo[e]; a[e + 8] = ahi[e]; }

            // ---- gates = h @ W_hh^T + xg : 8x V_WMMA_F32_16X16X32_F16
#pragma unroll
            for (int j = 0; j < 8; ++j) {
                acc[j] = __builtin_amdgcn_wmma_f32_16x16x32_f16(
                    false, a, false, btile[j], (short)0, acc[j], false, false);
            }

            // ---- Branch-free LSTM cell update; write h_t back to LDS as f16.
#pragma unroll
            for (int jp = 0; jp < 2; ++jp) {
#pragma unroll
                for (int r = 0; r < 8; ++r) {
                    const float ig = fast_sigmoid(acc[jp][r]);
                    const float fg = fast_sigmoid(acc[jp + 2][r]);
                    const float gg = fast_tanh(acc[jp + 4][r]);
                    const float og = fast_sigmoid(acc[jp + 6][r]);
                    const float c  = __builtin_fmaf(fg, cst[jp * 8 + r], ig * gg);
                    cst[jp * 8 + r] = c;
                    const float h = og * fast_tanh(c);
                    hl[(r + 8 * hi) * HID + 16 * jp + n] = (_Float16)h;
                }
            }
            __builtin_amdgcn_wave_barrier();  // keep store->next-load order
        }
    }

    // ---- Final FC: out[row] = h_T[row,:] . W_fc + b_fc   (OUTPUT = 1)
    if (lane < 16) {
        float s = b_fc[0];
#pragma unroll
        for (int k = 0; k < HID; ++k)
            s = __builtin_fmaf((float)hl[lane * HID + k], W_fc[k], s);
        out[rowbase + lane] = s;
    }
}

extern "C" void kernel_launch(void* const* d_in, const int* in_sizes, int n_in,
                              void* d_out, int out_size, void* d_ws, size_t ws_size,
                              hipStream_t stream) {
    const float* x    = (const float*)d_in[0];
    const float* W_ih = (const float*)d_in[1];
    const float* W_hh = (const float*)d_in[2];
    const float* b_ih = (const float*)d_in[3];
    const float* b_hh = (const float*)d_in[4];
    const float* W_fc = (const float*)d_in[5];
    const float* b_fc = (const float*)d_in[6];
    float* out = (float*)d_out;

    const int totalWaves = B_TOTAL / ROWS_PER_WAVE;          // 256
    const int blocks     = totalWaves / WAVES_PER_BLOCK;     // 32
    lstm_recur_kernel<<<blocks, WAVES_PER_BLOCK * 32, 0, stream>>>(
        x, W_ih, W_hh, b_ih, b_hh, W_fc, b_fc, out);
}
